// ContactGraspNet_61830349193995
// MI455X (gfx1250) — compile-verified, matrix operand order checked by reading, not run on hardware
//
#include <hip/hip_runtime.h>
#include <hip/hip_bf16.h>
#include <math.h>

// ContactGraspNet forward for gfx1250 (MI455X), round 3.
// Branch-free bf16 WMMA GEMMs:
//   - A staged bf16 [M][Kp] (Kp = align32, zero-padded by producers)
//   - W pre-packed transposed bf16 [Npad][Kp] (zero-padded)
//   - per-lane fragments = two contiguous 16B global_load_b128 each
//   - bias+BN folded into per-channel (scale, shift), zero in padding
// Main kernel computes a 32x32 tile per wave (2 A frags x 2 B frags ->
// 4 v_wmma per 8 b128 loads, 2x the wmma density of the 16x16 version);
// 16x16 variant retained for the narrow head-final layers (Npad=16).

typedef __attribute__((ext_vector_type(16))) __bf16 v16bf;
typedef __attribute__((ext_vector_type(8)))  __bf16 v8bf;
typedef __attribute__((ext_vector_type(8)))  float  v8f;

#define BN_EPS 1e-5f
#define NPTS   20000
#define NP1    2048
#define NP2    512

__device__ __forceinline__ v16bf frag_ld(const __bf16* p) {
  v8bf lo = *(const v8bf*)(p);       // K = base .. base+7
  v8bf hi = *(const v8bf*)(p + 16);  // K = base+16 .. base+23
  return __builtin_shufflevector(lo, hi, 0,1,2,3,4,5,6,7,8,9,10,11,12,13,14,15);
}

// ---------------- 32x32-tile WMMA GEMM (main path) --------------------------
__global__ void gemm32_bf16_wmma_kernel(const __bf16* __restrict__ A, int lda,
                                        const __bf16* __restrict__ Wt, int ldw,
                                        __bf16* __restrict__ C, int ldc,
                                        const float* __restrict__ sc,
                                        const float* __restrict__ sh,
                                        int Ksteps, int relu) {
  const int lane = threadIdx.x & 31;
  const int hi   = (lane >> 4) ? 8 : 0;   // lanes 16..31 -> K offset +8, M offset +8
  const int l16  = lane & 15;
  const int tn = blockIdx.x, tm = blockIdx.y;
  const __bf16* a0 = A  + (size_t)(tm * 32 + l16) * lda + hi;
  const __bf16* a1 = a0 + (size_t)16 * lda;
  const __bf16* b0 = Wt + (size_t)(tn * 32 + l16) * ldw + hi;
  const __bf16* b1 = b0 + (size_t)16 * ldw;
  v8f acc00 = {}, acc01 = {}, acc10 = {}, acc11 = {};
  for (int kk = 0; kk < Ksteps; ++kk) {
    v16bf af0 = frag_ld(a0);
    v16bf af1 = frag_ld(a1);
    v16bf bf0 = frag_ld(b0);
    v16bf bf1 = frag_ld(b1);
    acc00 = __builtin_amdgcn_wmma_f32_16x16x32_bf16(false, af0, false, bf0, (short)0, acc00, false, false);
    acc01 = __builtin_amdgcn_wmma_f32_16x16x32_bf16(false, af0, false, bf1, (short)0, acc01, false, false);
    acc10 = __builtin_amdgcn_wmma_f32_16x16x32_bf16(false, af1, false, bf0, (short)0, acc10, false, false);
    acc11 = __builtin_amdgcn_wmma_f32_16x16x32_bf16(false, af1, false, bf1, (short)0, acc11, false, false);
    a0 += 32; a1 += 32; b0 += 32; b1 += 32;
  }
  const int n0 = tn * 32 + l16;
  const int n1 = n0 + 16;
  const float s0 = sc[n0], t0 = sh[n0];
  const float s1 = sc[n1], t1 = sh[n1];
  __bf16* c0 = C + (size_t)(tm * 32 + hi) * ldc;        // rows for mi=0
  __bf16* c1 = C + (size_t)(tm * 32 + 16 + hi) * ldc;   // rows for mi=1
#pragma unroll
  for (int j = 0; j < 8; ++j) {
    float y00 = acc00[j] * s0 + t0;
    float y01 = acc01[j] * s1 + t1;
    float y10 = acc10[j] * s0 + t0;
    float y11 = acc11[j] * s1 + t1;
    if (relu) {
      y00 = fmaxf(y00, 0.0f); y01 = fmaxf(y01, 0.0f);
      y10 = fmaxf(y10, 0.0f); y11 = fmaxf(y11, 0.0f);
    }
    c0[(size_t)j * ldc + n0] = (__bf16)y00;
    c0[(size_t)j * ldc + n1] = (__bf16)y01;
    c1[(size_t)j * ldc + n0] = (__bf16)y10;
    c1[(size_t)j * ldc + n1] = (__bf16)y11;
  }
}

// ---------------- 16x16-tile WMMA GEMM (narrow head layers) -----------------
__global__ void gemm16_bf16_wmma_kernel(const __bf16* __restrict__ A, int lda,
                                        const __bf16* __restrict__ Wt, int ldw,
                                        __bf16* __restrict__ C, int ldc,
                                        const float* __restrict__ sc,
                                        const float* __restrict__ sh,
                                        int Ksteps, int relu) {
  const int lane = threadIdx.x & 31;
  const int hi   = (lane >> 4) ? 8 : 0;
  const int l16  = lane & 15;
  const int tn = blockIdx.x, tm = blockIdx.y;
  const __bf16* a0 = A  + (size_t)(tm * 16 + l16) * lda + hi;
  const __bf16* b0 = Wt + (size_t)(tn * 16 + l16) * ldw + hi;
  v8f acc = {};
  for (int kk = 0; kk < Ksteps; ++kk) {
    v16bf af = frag_ld(a0);
    v16bf bf = frag_ld(b0);
    acc = __builtin_amdgcn_wmma_f32_16x16x32_bf16(false, af, false, bf,
                                                  (short)0, acc, false, false);
    a0 += 32; b0 += 32;
  }
  const int n = tn * 16 + l16;
  const float s = sc[n], t = sh[n];
  __bf16* c0 = C + (size_t)(tm * 16 + hi) * ldc + n;
#pragma unroll
  for (int j = 0; j < 8; ++j) {
    float y = acc[j] * s + t;
    if (relu) y = fmaxf(y, 0.0f);
    c0[(size_t)j * ldc] = (__bf16)y;
  }
}

// ---------------- weight / epilogue packing ---------------------------------
__global__ void pack_w_kernel(const float* __restrict__ W, int K, int N,
                              __bf16* __restrict__ Wt, int Kp, int Npad) {
  size_t tid = (size_t)blockIdx.x * blockDim.x + threadIdx.x;
  if (tid >= (size_t)Npad * Kp) return;
  int n = (int)(tid / Kp), k = (int)(tid % Kp);
  float v = (k < K && n < N) ? W[(size_t)k * N + n] : 0.0f;
  Wt[tid] = (__bf16)v;
}

__global__ void pack_eff_kernel(const float* __restrict__ b,
                                const float* __restrict__ g,
                                const float* __restrict__ bt,
                                const float* __restrict__ m,
                                const float* __restrict__ v,
                                int N, int Npad, float* __restrict__ sc,
                                float* __restrict__ sh, int hasbn) {
  int n = blockIdx.x * blockDim.x + threadIdx.x;
  if (n >= Npad) return;
  if (n >= N) { sc[n] = 0.0f; sh[n] = 0.0f; return; }
  if (hasbn) {
    float s = rsqrtf(v[n] + BN_EPS) * g[n];
    sc[n] = s;
    sh[n] = (b[n] - m[n]) * s + bt[n];
  } else {
    sc[n] = 1.0f;
    sh[n] = b[n];
  }
}

// ---------------- pc [3,N] -> xyz0 [N,3] f32 (+ output slot) ----------------
__global__ void transpose_pc_kernel(const float* __restrict__ pc,
                                    float* __restrict__ xyz0,
                                    float* __restrict__ outxyz, int N) {
  int tid = blockIdx.x * blockDim.x + threadIdx.x;
  if (tid >= N * 3) return;
  int i = tid / 3, c = tid % 3;
  float v = pc[(size_t)c * N + i];
  xyz0[tid] = v;
  outxyz[tid] = v;
}

// ---------------- farthest point sampling (single block, serial) ------------
#define FPS_T 1024
__global__ void fps_kernel(const float* __restrict__ xyz, int N, int npoint,
                           int* __restrict__ outIdx, float* __restrict__ dist) {
  __shared__ float sval[FPS_T];
  __shared__ int   sidx[FPS_T];
  int t = threadIdx.x;
  for (int i = t; i < N; i += FPS_T) dist[i] = 1e10f;
  __syncthreads();
  int far = 0;
  for (int it = 0; it < npoint; ++it) {
    if (t == 0) outIdx[it] = far;
    float cx = xyz[far * 3 + 0], cy = xyz[far * 3 + 1], cz = xyz[far * 3 + 2];
    float best = -1.0f; int bidx = 0x7fffffff;
    for (int i = t; i < N; i += FPS_T) {
      float dx = xyz[i * 3 + 0] - cx;
      float dy = xyz[i * 3 + 1] - cy;
      float dz = xyz[i * 3 + 2] - cz;
      float d = dx * dx + dy * dy + dz * dz;
      float dd = fminf(dist[i], d);
      dist[i] = dd;
      if (dd > best) { best = dd; bidx = i; }
    }
    sval[t] = best; sidx[t] = bidx;
    __syncthreads();
    for (int s = FPS_T / 2; s > 0; s >>= 1) {
      if (t < s) {
        if (sval[t + s] > sval[t] ||
            (sval[t + s] == sval[t] && sidx[t + s] < sidx[t])) {
          sval[t] = sval[t + s]; sidx[t] = sidx[t + s];
        }
      }
      __syncthreads();
    }
    far = sidx[0];
    __syncthreads();
  }
}

__global__ void gather3_kernel(const float* __restrict__ xyz,
                               const int* __restrict__ idx,
                               float* __restrict__ out, int S) {
  int tid = blockIdx.x * blockDim.x + threadIdx.x;
  if (tid >= S * 3) return;
  out[tid] = xyz[(size_t)idx[tid / 3] * 3 + (tid % 3)];
}

// ---------------- ball query: first K indices with d^2 <= r^2 ---------------
__global__ void ball_idx_kernel(const float* __restrict__ xyz, int N,
                                const float* __restrict__ newxyz, int s0,
                                float r2, int K, int* __restrict__ out,
                                int Schunk) {
  int sg = blockIdx.x * blockDim.x + threadIdx.x;
  if (sg >= Schunk) return;
  int s = s0 + sg;
  float cx = newxyz[s * 3 + 0], cy = newxyz[s * 3 + 1], cz = newxyz[s * 3 + 2];
  int* o = out + (size_t)sg * K;
  int cnt = 0, first = 0;
  for (int i = 0; i < N && cnt < K; ++i) {
    float dx = xyz[i * 3 + 0] - cx;
    float dy = xyz[i * 3 + 1] - cy;
    float dz = xyz[i * 3 + 2] - cz;
    float d = dx * dx + dy * dy + dz * dz;
    if (d <= r2) { if (cnt == 0) first = i; o[cnt++] = i; }
  }
  for (; cnt < K; ++cnt) o[cnt] = first;
}

// ---------------- grouped gather into padded bf16 rows ----------------------
__global__ void group_gather_bf16_kernel(const float* __restrict__ xyz,
                                         const float* __restrict__ newxyz,
                                         const __bf16* __restrict__ points,
                                         int Cpts, const int* __restrict__ gi,
                                         int s0, int K, int rows, int cin,
                                         int Kp, __bf16* __restrict__ out) {
  size_t tid = (size_t)blockIdx.x * blockDim.x + threadIdx.x;
  if (tid >= (size_t)rows * Kp) return;
  int row = (int)(tid / Kp), c = (int)(tid % Kp);
  float v = 0.0f;
  if (c < cin) {
    int src = gi[row];
    int s = s0 + row / K;
    if (c < Cpts) v = (float)points[(size_t)src * Cpts + c];
    else v = xyz[(size_t)src * 3 + (c - Cpts)] - newxyz[(size_t)s * 3 + (c - Cpts)];
  }
  out[tid] = (__bf16)v;
}

// ---------------- max over K samples -> strided bf16 feature table ----------
__global__ void maxpool_bf16_kernel(const __bf16* __restrict__ in, int K, int C,
                                    __bf16* __restrict__ out, int Ctot,
                                    int chOff, int s0, int Schunk) {
  int tid = blockIdx.x * blockDim.x + threadIdx.x;
  if (tid >= Schunk * C) return;
  int sg = tid / C, c = tid % C;
  float m = -3.0e38f;
  const __bf16* p = in + (size_t)sg * K * C + c;
  for (int k = 0; k < K; ++k) m = fmaxf(m, (float)p[(size_t)k * C]);
  out[(size_t)(s0 + sg) * Ctot + chOff + c] = (__bf16)m;
}

__global__ void colmax_bf16_kernel(const __bf16* __restrict__ in, int Mrows,
                                   int C, __bf16* __restrict__ out) {
  int c = blockIdx.x * blockDim.x + threadIdx.x;
  if (c >= C) return;
  float m = -3.0e38f;
  for (int r = 0; r < Mrows; ++r) m = fmaxf(m, (float)in[(size_t)r * C + c]);
  out[c] = (__bf16)m;
}

// ---------------- block copies / broadcasts / padding -----------------------
__global__ void copy_bf16_block_kernel(const __bf16* __restrict__ src, int Csrc,
                                       __bf16* __restrict__ dst, int Ctot,
                                       int chOff, int M) {
  size_t tid = (size_t)blockIdx.x * blockDim.x + threadIdx.x;
  if (tid >= (size_t)M * Csrc) return;
  int r = (int)(tid / Csrc), c = (int)(tid % Csrc);
  dst[(size_t)r * Ctot + chOff + c] = src[tid];
}

__global__ void copy_f32_block_bf16_kernel(const float* __restrict__ src,
                                           int Csrc, __bf16* __restrict__ dst,
                                           int Ctot, int chOff, int M) {
  size_t tid = (size_t)blockIdx.x * blockDim.x + threadIdx.x;
  if (tid >= (size_t)M * Csrc) return;
  int r = (int)(tid / Csrc), c = (int)(tid % Csrc);
  dst[(size_t)r * Ctot + chOff + c] = (__bf16)src[tid];
}

__global__ void broadcast_row_bf16_kernel(const __bf16* __restrict__ src,
                                          int Csrc, __bf16* __restrict__ dst,
                                          int Ctot, int chOff, int M) {
  size_t tid = (size_t)blockIdx.x * blockDim.x + threadIdx.x;
  if (tid >= (size_t)M * Csrc) return;
  int r = (int)(tid / Csrc), c = (int)(tid % Csrc);
  dst[(size_t)r * Ctot + chOff + c] = src[c];
}

__global__ void zero_cols_bf16_kernel(__bf16* __restrict__ dst, int Ctot,
                                      int c0, int M) {
  int Cz = Ctot - c0;
  size_t tid = (size_t)blockIdx.x * blockDim.x + threadIdx.x;
  if (tid >= (size_t)M * Cz) return;
  int r = (int)(tid / Cz), c = c0 + (int)(tid % Cz);
  dst[(size_t)r * Ctot + c] = (__bf16)0.0f;
}

// ---------------- 3-NN inverse-distance interpolation (bf16 feats) ----------
__global__ void interp3_bf16_kernel(const float* __restrict__ q, int Nq,
                                    const float* __restrict__ ref, int S,
                                    const __bf16* __restrict__ feats, int C,
                                    __bf16* __restrict__ out, int Ctot,
                                    int chOff) {
  int i = blockIdx.x * blockDim.x + threadIdx.x;
  if (i >= Nq) return;
  float qx = q[i * 3 + 0], qy = q[i * 3 + 1], qz = q[i * 3 + 2];
  float d0 = 3e38f, d1 = 3e38f, d2 = 3e38f;
  int i0 = 0, i1 = 0, i2 = 0;
  for (int s = 0; s < S; ++s) {
    float dx = ref[s * 3 + 0] - qx;
    float dy = ref[s * 3 + 1] - qy;
    float dz = ref[s * 3 + 2] - qz;
    float d = dx * dx + dy * dy + dz * dz;
    if (d < d0)      { d2 = d1; i2 = i1; d1 = d0; i1 = i0; d0 = d; i0 = s; }
    else if (d < d1) { d2 = d1; i2 = i1; d1 = d; i1 = s; }
    else if (d < d2) { d2 = d; i2 = s; }
  }
  float w0 = 1.0f / (d0 + 1e-8f);
  float w1 = 1.0f / (d1 + 1e-8f);
  float w2 = 1.0f / (d2 + 1e-8f);
  float ws = w0 + w1 + w2;
  w0 /= ws; w1 /= ws; w2 /= ws;
  const __bf16* f0 = feats + (size_t)i0 * C;
  const __bf16* f1 = feats + (size_t)i1 * C;
  const __bf16* f2 = feats + (size_t)i2 * C;
  __bf16* o = out + (size_t)i * Ctot + chOff;
  for (int c = 0; c < C; ++c)
    o[c] = (__bf16)(w0 * (float)f0[c] + w1 * (float)f1[c] + w2 * (float)f2[c]);
}

// ---------------- head epilogues (bf16 in, f32 out) -------------------------
__global__ void l2norm3_kernel(const __bf16* __restrict__ in, int ld,
                               float* __restrict__ out, int Npts) {
  int i = blockIdx.x * blockDim.x + threadIdx.x;
  if (i >= Npts) return;
  float x = (float)in[(size_t)i * ld + 0];
  float y = (float)in[(size_t)i * ld + 1];
  float z = (float)in[(size_t)i * ld + 2];
  float n = fmaxf(sqrtf(x * x + y * y + z * z), 1e-12f);
  out[i * 3 + 0] = x / n; out[i * 3 + 1] = y / n; out[i * 3 + 2] = z / n;
}

__global__ void ortho_norm_kernel(const __bf16* __restrict__ ap, int ld,
                                  const float* __restrict__ gdn,
                                  float* __restrict__ out, int Npts) {
  int i = blockIdx.x * blockDim.x + threadIdx.x;
  if (i >= Npts) return;
  float ax = (float)ap[(size_t)i * ld + 0];
  float ay = (float)ap[(size_t)i * ld + 1];
  float az = (float)ap[(size_t)i * ld + 2];
  float gx = gdn[i * 3 + 0], gy = gdn[i * 3 + 1], gz = gdn[i * 3 + 2];
  float d = ax * gx + ay * gy + az * gz;
  float vx = ax - d * gx, vy = ay - d * gy, vz = az - d * gz;
  float n = fmaxf(sqrtf(vx * vx + vy * vy + vz * vz), 1e-12f);
  out[i * 3 + 0] = vx / n; out[i * 3 + 1] = vy / n; out[i * 3 + 2] = vz / n;
}

__global__ void raw_sig_kernel(const __bf16* __restrict__ in, int ld, int C,
                               float* __restrict__ raw, float* __restrict__ sig,
                               int Npts) {
  int tid = blockIdx.x * blockDim.x + threadIdx.x;
  if (tid >= Npts * C) return;
  int i = tid / C, c = tid % C;
  float x = (float)in[(size_t)i * ld + c];
  raw[tid] = x;
  sig[tid] = 1.0f / (1.0f + expf(-x));
}

// =============================================================================
struct ConvP { const float *w, *b, *g, *bt, *m, *v; int cin, cout; };
struct Packed { __bf16* wt; float* sc; float* sh; int Kp, Npad, cout; };

static inline int ceil_div(int a, int b) { return (a + b - 1) / b; }
static inline int align_up(int a, int b) { return ceil_div(a, b) * b; }

extern "C" void kernel_launch(void* const* d_in, const int* in_sizes, int n_in,
                              void* d_out, int out_size, void* d_ws, size_t ws_size,
                              hipStream_t stream) {
  (void)in_sizes; (void)n_in; (void)out_size; (void)ws_size;

  // ---- parse params (setup_inputs dict order; per-conv order w,b,g,bt,m,v) --
  const float* pc = (const float*)d_in[0];
  int ii = 1;
  auto mk = [&](int cin, int cout) -> ConvP {
    ConvP p;
    p.w = (const float*)d_in[ii++]; p.b  = (const float*)d_in[ii++];
    p.g = (const float*)d_in[ii++]; p.bt = (const float*)d_in[ii++];
    p.m = (const float*)d_in[ii++]; p.v  = (const float*)d_in[ii++];
    p.cin = cin; p.cout = cout; return p;
  };
  auto mklin = [&](int cin, int cout) -> ConvP {
    ConvP p;
    p.w = (const float*)d_in[ii++]; p.b = (const float*)d_in[ii++];
    p.g = p.bt = p.m = p.v = nullptr;
    p.cin = cin; p.cout = cout; return p;
  };

  const int MLP1[3][3] = {{32, 32, 64}, {64, 64, 128}, {64, 96, 128}};
  const int MLP2[3][3] = {{64, 64, 128}, {128, 128, 256}, {128, 128, 256}};
  ConvP sa1[3][3], sa2[3][3], sa3[3], fp1[2], fp2[2], fp3[2];
  for (int b = 0; b < 3; ++b) { int cin = 3;
    for (int l = 0; l < 3; ++l) { sa1[b][l] = mk(cin, MLP1[b][l]); cin = MLP1[b][l]; } }
  for (int b = 0; b < 3; ++b) { int cin = 323;
    for (int l = 0; l < 3; ++l) { sa2[b][l] = mk(cin, MLP2[b][l]); cin = MLP2[b][l]; } }
  { int cin = 643; const int d[3] = {256, 512, 640};
    for (int l = 0; l < 3; ++l) { sa3[l] = mk(cin, d[l]); cin = d[l]; } }
  fp1[0] = mk(1280, 256); fp1[1] = mk(256, 256);
  fp2[0] = mk(576, 256);  fp2[1] = mk(256, 128);
  fp3[0] = mk(131, 128);  fp3[1] = mk(128, 128);
  ConvP fc1     = mk(128, 128), fc3     = mklin(128, 3);
  ConvP fc1_app = mk(128, 128), fc3_app = mklin(128, 3);
  ConvP fc1_off = mk(128, 128), fc2_off = mklin(128, 10);
  ConvP fc1_seg = mk(128, 128), fc2_seg = mklin(128, 1);

  // ---- workspace bump allocator (64B aligned blocks) ------------------------
  char* base = (char*)d_ws;
  size_t wo = 0;
  auto allocB = [&](size_t bytes) -> void* {
    wo = (wo + 63) & ~(size_t)63;
    void* p = base + wo;
    wo += bytes;
    return p;
  };
  auto allocF  = [&](size_t n) { return (float*)allocB(n * 4); };
  auto allocH  = [&](size_t n) { return (__bf16*)allocB(n * 2); };
  auto allocI  = [&](size_t n) { return (int*)allocB(n * 4); };
  auto ew = [&](size_t total) { return dim3((unsigned)((total + 255) / 256)); };

  // ---- pack all layer weights into bf16 transposed + fused epilogue params --
  auto packL = [&](const ConvP& p) -> Packed {
    Packed q;
    q.Kp = align_up(p.cin, 32);
    q.Npad = align_up(p.cout, 16);
    q.cout = p.cout;
    q.wt = allocH((size_t)q.Npad * q.Kp);
    q.sc = allocF(q.Npad);
    q.sh = allocF(q.Npad);
    pack_w_kernel<<<ew((size_t)q.Npad * q.Kp), 256, 0, stream>>>(
        p.w, p.cin, p.cout, q.wt, q.Kp, q.Npad);
    pack_eff_kernel<<<ew(q.Npad), 256, 0, stream>>>(
        p.b, p.g, p.bt, p.m, p.v, p.cout, q.Npad, q.sc, q.sh, p.g != nullptr);
    return q;
  };
  Packed P_sa1[3][3], P_sa2[3][3], P_sa3[3], P_fp1[2], P_fp2[2], P_fp3[2];
  for (int b = 0; b < 3; ++b) for (int l = 0; l < 3; ++l) P_sa1[b][l] = packL(sa1[b][l]);
  for (int b = 0; b < 3; ++b) for (int l = 0; l < 3; ++l) P_sa2[b][l] = packL(sa2[b][l]);
  for (int l = 0; l < 3; ++l) P_sa3[l] = packL(sa3[l]);
  for (int l = 0; l < 2; ++l) { P_fp1[l] = packL(fp1[l]); P_fp2[l] = packL(fp2[l]); P_fp3[l] = packL(fp3[l]); }
  Packed P_fc1 = packL(fc1), P_fc3 = packL(fc3);
  Packed P_fc1a = packL(fc1_app), P_fc3a = packL(fc3_app);
  Packed P_fc1o = packL(fc1_off), P_fc2o = packL(fc2_off);
  Packed P_fc1s = packL(fc1_seg), P_fc2s = packL(fc2_seg);

  // ---- persistent activations (coords f32, features bf16) -------------------
  float*  xyz0 = allocF((size_t)NPTS * 3);
  float*  l1x  = allocF((size_t)NP1 * 3);
  float*  l2x  = allocF((size_t)NP2 * 3);
  __bf16* l1p  = allocH((size_t)NP1 * 320);
  __bf16* l2p  = allocH((size_t)NP2 * 640);
  __bf16* l3p  = allocH(640);
  __bf16* l2pf = allocH((size_t)NP2 * 256);
  __bf16* l1pf = allocH((size_t)NP1 * 128);
  __bf16* feat = allocH((size_t)NPTS * 128);
  float*  fdist = allocF(NPTS);
  int* fidx1 = allocI(NP1);
  int* fidx2 = allocI(NP2);
  int* ballidx = allocI(32768);
  __bf16* bufA = allocH(5000000);
  __bf16* bufB = allocH(5000000);

  float* out = (float*)d_out;
  float* o_gdn    = out;             // 60000
  float* o_segraw = out + 60000;     // 20000
  float* o_segsig = out + 80000;     // 20000
  float* o_offraw = out + 100000;    // 200000
  float* o_offsig = out + 300000;    // 200000
  float* o_apo    = out + 500000;    // 60000
  float* o_xyz    = out + 560000;    // 60000

  auto gemm = [&](const __bf16* A, int lda, const Packed& L, __bf16* C, int ldc,
                  int Mrows, int relu) {
    if ((Mrows % 32) == 0 && (L.Npad % 32) == 0) {
      dim3 grid(L.Npad / 32, Mrows / 32);
      gemm32_bf16_wmma_kernel<<<grid, dim3(32), 0, stream>>>(
          A, lda, L.wt, L.Kp, C, ldc, L.sc, L.sh, L.Kp / 32, relu);
    } else {
      dim3 grid(L.Npad / 16, Mrows / 16);
      gemm16_bf16_wmma_kernel<<<grid, dim3(32), 0, stream>>>(
          A, lda, L.wt, L.Kp, C, ldc, L.sc, L.sh, L.Kp / 32, relu);
    }
  };
  // chain: each layer's output stride == its Npad == next layer's Kp
  auto run_chain = [&](const __bf16* in, int Mrows, const Packed* ch, int nlayers) -> __bf16* {
    const __bf16* cur = in;
    int lda = ch[0].Kp;
    __bf16* dst = (in == bufA) ? bufB : bufA;
    for (int l = 0; l < nlayers; ++l) {
      gemm(cur, lda, ch[l], dst, ch[l].Npad, Mrows, 1);
      cur = dst;
      lda = ch[l].Npad;
      dst = (dst == bufA) ? bufB : bufA;
    }
    return (__bf16*)cur;
  };

  // ---- xyz0 = pc^T ----------------------------------------------------------
  transpose_pc_kernel<<<ew((size_t)NPTS * 3), 256, 0, stream>>>(pc, xyz0, o_xyz, NPTS);

  // ---- SA1 ------------------------------------------------------------------
  fps_kernel<<<1, FPS_T, 0, stream>>>(xyz0, NPTS, NP1, fidx1, fdist);
  gather3_kernel<<<ew((size_t)NP1 * 3), 256, 0, stream>>>(xyz0, fidx1, l1x, NP1);
  {
    const float R1[3] = {0.1f, 0.2f, 0.4f};
    const int  K1[3]  = {32, 64, 128};
    const int  OFF1[3] = {0, 64, 192};   // couts 64,128,128 -> 320
    const int SC = 256;
    for (int br = 0; br < 3; ++br) {
      int K = K1[br];
      float r2 = R1[br] * R1[br];
      int Kp = P_sa1[br][0].Kp;  // 32
      for (int s0 = 0; s0 < NP1; s0 += SC) {
        int Sc = SC;
        int rows = Sc * K;
        ball_idx_kernel<<<ew(Sc), 256, 0, stream>>>(xyz0, NPTS, l1x, s0, r2, K, ballidx, Sc);
        group_gather_bf16_kernel<<<ew((size_t)rows * Kp), 256, 0, stream>>>(
            xyz0, l1x, nullptr, 0, ballidx, s0, K, rows, 3, Kp, bufA);
        __bf16* fin = run_chain(bufA, rows, P_sa1[br], 3);
        int C = P_sa1[br][2].cout;
        maxpool_bf16_kernel<<<ew((size_t)Sc * C), 256, 0, stream>>>(
            fin, K, C, l1p, 320, OFF1[br], s0, Sc);
      }
    }
  }

  // ---- SA2 ------------------------------------------------------------------
  fps_kernel<<<1, FPS_T, 0, stream>>>(l1x, NP1, NP2, fidx2, fdist);
  gather3_kernel<<<ew((size_t)NP2 * 3), 256, 0, stream>>>(l1x, fidx2, l2x, NP2);
  {
    const float R2[3] = {0.2f, 0.4f, 0.8f};
    const int  K2[3]  = {32, 64, 128};
    const int  OFF2[3] = {0, 128, 384};  // couts 128,256,256 -> 640
    const int SC = 64;
    for (int br = 0; br < 3; ++br) {
      int K = K2[br];
      float r2 = R2[br] * R2[br];
      int Kp = P_sa2[br][0].Kp;  // 352
      for (int s0 = 0; s0 < NP2; s0 += SC) {
        int Sc = SC;
        int rows = Sc * K;
        ball_idx_kernel<<<ew(Sc), 256, 0, stream>>>(l1x, NP1, l2x, s0, r2, K, ballidx, Sc);
        group_gather_bf16_kernel<<<ew((size_t)rows * Kp), 256, 0, stream>>>(
            l1x, l2x, l1p, 320, ballidx, s0, K, rows, 323, Kp, bufA);
        __bf16* fin = run_chain(bufA, rows, P_sa2[br], 3);
        int C = P_sa2[br][2].cout;
        maxpool_bf16_kernel<<<ew((size_t)Sc * C), 256, 0, stream>>>(
            fin, K, C, l2p, 640, OFF2[br], s0, Sc);
      }
    }
  }

  // ---- SA3 (group_all): rows [512, Kp=672] = [l2x | l2p | 0pad] -------------
  {
    int Kp = P_sa3[0].Kp;  // 672
    copy_f32_block_bf16_kernel<<<ew((size_t)NP2 * 3), 256, 0, stream>>>(l2x, 3, bufA, Kp, 0, NP2);
    copy_bf16_block_kernel<<<ew((size_t)NP2 * 640), 256, 0, stream>>>(l2p, 640, bufA, Kp, 3, NP2);
    zero_cols_bf16_kernel<<<ew((size_t)NP2 * (Kp - 643)), 256, 0, stream>>>(bufA, Kp, 643, NP2);
    __bf16* fin = run_chain(bufA, NP2, P_sa3, 3);
    colmax_bf16_kernel<<<ew(640), 256, 0, stream>>>(fin, NP2, 640, l3p);
  }

  // ---- FP1 (S==1 broadcast): rows [512, 1280] -> l2pf [512,256] -------------
  {
    copy_bf16_block_kernel<<<ew((size_t)NP2 * 640), 256, 0, stream>>>(l2p, 640, bufA, 1280, 0, NP2);
    broadcast_row_bf16_kernel<<<ew((size_t)NP2 * 640), 256, 0, stream>>>(l3p, 640, bufA, 1280, 640, NP2);
    __bf16* fin = run_chain(bufA, NP2, P_fp1, 2);
    copy_bf16_block_kernel<<<ew((size_t)NP2 * 256), 256, 0, stream>>>(fin, 256, l2pf, 256, 0, NP2);
  }

  // ---- FP2: rows [2048, 576] -> l1pf [2048,128] -----------------------------
  {
    copy_bf16_block_kernel<<<ew((size_t)NP1 * 320), 256, 0, stream>>>(l1p, 320, bufA, 576, 0, NP1);
    interp3_bf16_kernel<<<ew(NP1), 256, 0, stream>>>(l1x, NP1, l2x, NP2, l2pf, 256, bufA, 576, 320);
    __bf16* fin = run_chain(bufA, NP1, P_fp2, 2);
    copy_bf16_block_kernel<<<ew((size_t)NP1 * 128), 256, 0, stream>>>(fin, 128, l1pf, 128, 0, NP1);
  }

  // ---- FP3: rows [20000, Kp=160] -> feat [20000,128] ------------------------
  {
    int Kp = P_fp3[0].Kp;  // 160
    copy_f32_block_bf16_kernel<<<ew((size_t)NPTS * 3), 256, 0, stream>>>(xyz0, 3, bufA, Kp, 0, NPTS);
    interp3_bf16_kernel<<<ew(NPTS), 256, 0, stream>>>(xyz0, NPTS, l1x, NP1, l1pf, 128, bufA, Kp, 3);
    zero_cols_bf16_kernel<<<ew((size_t)NPTS * (Kp - 131)), 256, 0, stream>>>(bufA, Kp, 131, NPTS);
    __bf16* fin = run_chain(bufA, NPTS, P_fp3, 2);
    copy_bf16_block_kernel<<<ew((size_t)NPTS * 128), 256, 0, stream>>>(fin, 128, feat, 128, 0, NPTS);
  }

  // ---- heads ----------------------------------------------------------------
  // grasp direction -> gdn (first; approach head depends on it)
  gemm(feat, 128, P_fc1, bufA, 128, NPTS, 1);
  gemm(bufA, 128, P_fc3, bufB, 16, NPTS, 0);
  l2norm3_kernel<<<ew(NPTS), 256, 0, stream>>>(bufB, 16, o_gdn, NPTS);
  // approach -> Gram-Schmidt vs gdn, normalize
  gemm(feat, 128, P_fc1a, bufA, 128, NPTS, 1);
  gemm(bufA, 128, P_fc3a, bufB, 16, NPTS, 0);
  ortho_norm_kernel<<<ew(NPTS), 256, 0, stream>>>(bufB, 16, o_gdn, o_apo, NPTS);
  // grasp offset bins (raw + sigmoid)
  gemm(feat, 128, P_fc1o, bufA, 128, NPTS, 1);
  gemm(bufA, 128, P_fc2o, bufB, 16, NPTS, 0);
  raw_sig_kernel<<<ew((size_t)NPTS * 10), 256, 0, stream>>>(bufB, 16, 10, o_offraw, o_offsig, NPTS);
  // segmentation (raw + sigmoid)
  gemm(feat, 128, P_fc1s, bufA, 128, NPTS, 1);
  gemm(bufA, 128, P_fc2s, bufB, 16, NPTS, 0);
  raw_sig_kernel<<<ew(NPTS), 256, 0, stream>>>(bufB, 16, 1, o_segraw, o_segsig, NPTS);
}
